// SliceAttention_76459007803677
// MI455X (gfx1250) — matrix operand order repaired and checked by reading, not compiled, not run on hardware
//
#include <hip/hip_runtime.h>
#include <cstdint>
#include <cstddef>

// ---------------------------------------------------------------------------
// SliceAttention on MI455X (gfx1250, wave32, WMMA).
// All heavy matmuls use v_wmma_f32_16x16x32_bf16; softmax/norm math in f32.
// GEMM staging uses GLOBAL_LOAD_ASYNC_TO_LDS_B128 (ASYNCcnt) when available,
// with double-buffered LDS tiles to overlap loads with WMMA issue.
// ---------------------------------------------------------------------------

#define DEV __device__ __forceinline__

typedef __attribute__((ext_vector_type(16))) __bf16 v16bf;
typedef __attribute__((ext_vector_type(8)))  float  v8f;
typedef __attribute__((ext_vector_type(4)))  int    v4i;

static constexpr int Bb = 4;      // batch
static constexpr int Nn = 8192;   // tokens
static constexpr int Dd = 1024;   // model dim
static constexpr int Hh = 16;     // heads
static constexpr int Mm = 64;     // slices
static constexpr int HD = 64;     // head dim

#if defined(__gfx1250__) && __has_builtin(__builtin_amdgcn_global_load_async_to_lds_b128)
#define USE_ASYNC_LDS 1
#else
#define USE_ASYNC_LDS 0
#endif

DEV unsigned short f2bf(float f) {
  union { float f; unsigned int u; } x; x.f = f;
  unsigned int r = x.u + 0x7FFFu + ((x.u >> 16) & 1u);   // round-to-nearest-even
  return (unsigned short)(r >> 16);
}
DEV float bf2f(unsigned short u) {
  union { unsigned int u; float f; } x; x.u = ((unsigned int)u) << 16;
  return x.f;
}

// Fragment = 16 bf16 per lane = two 16B chunks at ushort offsets off0/off1.
DEV v16bf load_frag(const unsigned short* p, int off0, int off1) {
  union { v16bf v; uint4 q[2]; } u;
  u.q[0] = *(const uint4*)(p + off0);
  u.q[1] = *(const uint4*)(p + off1);
  return u.v;
}

DEV v8f zero8() {
  v8f z;
#pragma unroll
  for (int i = 0; i < 8; ++i) z[i] = 0.0f;
  return z;
}

DEV v8f wmma_bf16(v16bf a, v16bf b, v8f c) {
  // D = A(16x32) x B(32x16) + C, f32 accumulate
  return __builtin_amdgcn_wmma_f32_16x16x32_bf16(false, a, false, b, (short)0, c,
                                                 false, false);
}

// Wait until ASYNCcnt <= N (no-op when async path unused).
template <int N>
DEV void wait_async() {
#if USE_ASYNC_LDS
#if __has_builtin(__builtin_amdgcn_s_wait_asynccnt)
  __builtin_amdgcn_s_wait_asynccnt(N);
#else
  asm volatile("s_wait_asynccnt %0" ::"i"(N) : "memory");
#endif
#endif
}

// One 16-byte global->LDS transfer (async when available).
DEV void copy_b128(const unsigned short* g, unsigned short* l) {
#if USE_ASYNC_LDS
  __builtin_amdgcn_global_load_async_to_lds_b128(
      (__attribute__((address_space(1))) v4i*)g,
      (__attribute__((address_space(3))) v4i*)l, 0, 0);
#else
  *(uint4*)l = *(const uint4*)g;
#endif
}

// ---------------------------------------------------------------------------
// elementwise helpers
// ---------------------------------------------------------------------------
__global__ void k_f32_to_bf16(const float* __restrict__ in,
                              unsigned short* __restrict__ out, int n) {
  int i = blockIdx.x * blockDim.x + threadIdx.x;
  if (i < n) out[i] = f2bf(in[i]);
}

__global__ void k_zero_f32(float* __restrict__ p, int n) {
  int i = blockIdx.x * blockDim.x + threadIdx.x;
  if (i < n) p[i] = 0.0f;
}

// ---------------------------------------------------------------------------
// Generic GEMM: C[M,Nc] = A[M,K](bf16) x W[Nc,K]^T(bf16) + bias.
// Block = 256 threads (8 waves), 128x64 output tile, K-step 32.
// Double-buffered LDS tiles; staging = 3 async b128 per thread per step
// (A: 2, W: 1), s_wait_asynccnt 3 overlaps slab k+1's loads with slab k's
// WMMAs. Wave w owns rows w*16..w*16+15 and all four 16-col tiles.
// ---------------------------------------------------------------------------
template <bool BF16OUT>
__global__ void k_gemm(const unsigned short* __restrict__ A,
                       const unsigned short* __restrict__ W,
                       const float* __restrict__ bias,
                       void* __restrict__ Cout, int Nc, int K) {
  __shared__ unsigned short At[2][128 * 32];
  __shared__ unsigned short Bt[2][64 * 32];
  const int n0 = blockIdx.x * 64, m0 = blockIdx.y * 128;
  const int tid = threadIdx.x, wave = tid >> 5, lane = tid & 31;
  const int c0 = (lane >> 4) * 8;    // A-matrix K-chunk select per ISA layout
  const int c0b = (lane >> 4) * 16;  // unused; B fragments load with c0, same k order as A

  // staging assignment (uniform across all 256 threads / 8 waves)
  const int arow = tid >> 1, akoff = (tid & 1) * 16;  // A: 128 rows, 2 x b128
  const int brow = tid >> 2, bkoff = (tid & 3) * 8;   // W: 64 rows, 1 x b128

  auto stage = [&](int buf, int k0) {
    const unsigned short* ga = A + (size_t)(m0 + arow) * K + k0 + akoff;
    unsigned short* la = &At[buf][arow * 32 + akoff];
    copy_b128(ga, la);
    copy_b128(ga + 8, la + 8);
    const unsigned short* gw = W + (size_t)(n0 + brow) * K + k0 + bkoff;
    copy_b128(gw, &Bt[buf][brow * 32 + bkoff]);
  };

  v8f acc[4];
#pragma unroll
  for (int j = 0; j < 4; ++j) acc[j] = zero8();

  stage(0, 0);
  int buf = 0;
  for (int k0 = 0; k0 < K; k0 += 32) {
    if (k0 + 32 < K) {
      stage(buf ^ 1, k0 + 32);
      wait_async<3>();   // current slab done; next slab (3 asyncs) in flight
    } else {
      wait_async<0>();
    }
    __syncthreads();

    v16bf a = load_frag(&At[buf][(wave * 16 + (lane & 15)) * 32], c0, c0 + 16);
#pragma unroll
    for (int j = 0; j < 4; ++j) {
      v16bf b = load_frag(&Bt[buf][(j * 16 + (lane & 15)) * 32], c0, c0 + 16);
      acc[j] = wmma_bf16(a, b, acc[j]);
    }
    __syncthreads();
    buf ^= 1;
  }

  const int hi = (lane >> 4) * 8;
#pragma unroll
  for (int j = 0; j < 4; ++j) {
    const int col = n0 + j * 16 + (lane & 15);
    const float bv = bias ? bias[col] : 0.0f;
#pragma unroll
    for (int v = 0; v < 8; ++v) {
      const size_t idx = (size_t)(m0 + wave * 16 + v + hi) * Nc + col;
      const float val = acc[j][v] + bv;
      if (BF16OUT) ((unsigned short*)Cout)[idx] = f2bf(val);
      else         ((float*)Cout)[idx] = val;
    }
  }
}

// ---------------------------------------------------------------------------
// Slice scores + softmax over M.
// One wave per (b,h,16-token tile). S[m,n] = wtq[h,m,:] . wtk[b,h,n,:],
// computed as 4 row-tiles x 2 K-steps of WMMA straight from global (both
// operands are K-contiguous). Softmax over the 64 m values (32 regs per lane
// + shfl_xor(16) to merge the two lane halves). Stores bf16 slice weights.
// ---------------------------------------------------------------------------
__global__ void k_scores_softmax(const unsigned short* __restrict__ wtqb,
                                 const unsigned short* __restrict__ wtkb,
                                 const float* __restrict__ temp,
                                 unsigned short* __restrict__ wts) {
  const int n0 = blockIdx.x * 16, h = blockIdx.y, b = blockIdx.z;
  const int lane = threadIdx.x & 31;
  const int c0 = (lane >> 4) * 8, c0b = (lane >> 4) * 16;

  v8f acc[4];
#pragma unroll
  for (int t = 0; t < 4; ++t) acc[t] = zero8();

  const unsigned short* bbase =
      wtkb + (size_t)(b * Nn + n0 + (lane & 15)) * Dd + h * HD;
  const unsigned short* abase = wtqb + h * (Mm * HD);

#pragma unroll
  for (int k0 = 0; k0 < HD; k0 += 32) {
    v16bf bfr = load_frag(bbase + k0, c0, c0 + 16);
#pragma unroll
    for (int t = 0; t < 4; ++t) {
      v16bf a = load_frag(abase + (t * 16 + (lane & 15)) * HD + k0, c0, c0 + 16);
      acc[t] = wmma_bf16(a, bfr, acc[t]);
    }
  }

  const float invt = 1.0f / temp[h];
  float s[4][8];
  float mx = -3.0e38f;
#pragma unroll
  for (int t = 0; t < 4; ++t)
#pragma unroll
    for (int v = 0; v < 8; ++v) {
      s[t][v] = acc[t][v] * invt;
      mx = fmaxf(mx, s[t][v]);
    }
  mx = fmaxf(mx, __shfl_xor(mx, 16, 32));
  float sum = 0.0f;
#pragma unroll
  for (int t = 0; t < 4; ++t)
#pragma unroll
    for (int v = 0; v < 8; ++v) {
      const float e = __expf(s[t][v] - mx);
      s[t][v] = e;
      sum += e;
    }
  sum += __shfl_xor(sum, 16, 32);
  const float inv = 1.0f / sum;

  const int hi = (lane >> 4) * 8;
  unsigned short* wbase =
      wts + (size_t)((b * Hh + h) * Mm) * Nn + n0 + (lane & 15);
#pragma unroll
  for (int t = 0; t < 4; ++t)
#pragma unroll
    for (int v = 0; v < 8; ++v) {
      const int m = t * 16 + v + hi;
      wbase[(size_t)m * Nn] = f2bf(s[t][v] * inv);
    }
}

// slice_norm[b,h,m] = sum_n weights[b,h,m,n]
__global__ void k_slice_norm(const unsigned short* __restrict__ wts,
                             float* __restrict__ snorm) {
  __shared__ float sd[256];
  const int row = blockIdx.x;  // (b*H+h)*M + m
  const unsigned short* p = wts + (size_t)row * Nn;
  float sum = 0.0f;
  for (int n = threadIdx.x; n < Nn; n += 256) sum += bf2f(p[n]);
  sd[threadIdx.x] = sum;
  __syncthreads();
  for (int s = 128; s > 0; s >>= 1) {
    if (threadIdx.x < (unsigned)s) sd[threadIdx.x] += sd[threadIdx.x + s];
    __syncthreads();
  }
  if (threadIdx.x == 0) snorm[row] = sd[0];
}

// ---------------------------------------------------------------------------
// qkv_tok[b,h,m,0:192] = sum_n weights[b,h,m,n] * qkv[b,n, h*192 + d]
// Grid (B*H, 8 K-chunks of 1024 tokens). 256 threads = 8 waves: wave w owns
// m-rows (w&3)*16 and 6 of 12 d-tiles. qkv K-slab is transposed into LDS so
// B-fragments are K-contiguous. Partial results accumulated via f32 atomics.
// ---------------------------------------------------------------------------
__global__ void k_pool_qkv(const unsigned short* __restrict__ wts,
                           const unsigned short* __restrict__ qkvb,
                           float* __restrict__ qtok) {
  __shared__ unsigned short ldsT[192 * 40];  // [d][n_local], padded rows
  const int bh = blockIdx.x, b = bh >> 4, h = bh & 15;
  const int nbase = blockIdx.y * 1024;
  const int tid = threadIdx.x, wave = tid >> 5, lane = tid & 31;
  const int r = wave & 3, g = wave >> 2;
  const int c0 = (lane >> 4) * 8, c0b = (lane >> 4) * 16;
  const int n_l = tid >> 3, d0 = (tid & 7) * 24;

  v8f acc[6];
#pragma unroll
  for (int j = 0; j < 6; ++j) acc[j] = zero8();

  for (int nk = 0; nk < 1024; nk += 32) {
    __syncthreads();
    // stage 32 tokens x 192 dims, transposed into LDS
    const unsigned short* src =
        qkvb + (size_t)(b * Nn + nbase + nk + n_l) * (3 * Dd) + h * 192 + d0;
    uint4 u0 = *(const uint4*)src;
    uint4 u1 = *(const uint4*)(src + 8);
    uint4 u2 = *(const uint4*)(src + 16);
    const unsigned short* p0 = (const unsigned short*)&u0;
    const unsigned short* p1 = (const unsigned short*)&u1;
    const unsigned short* p2 = (const unsigned short*)&u2;
#pragma unroll
    for (int j = 0; j < 8; ++j) {
      ldsT[(d0 + j) * 40 + n_l]      = p0[j];
      ldsT[(d0 + 8 + j) * 40 + n_l]  = p1[j];
      ldsT[(d0 + 16 + j) * 40 + n_l] = p2[j];
    }
    __syncthreads();

    v16bf a = load_frag(
        wts + (size_t)(bh * Mm + r * 16 + (lane & 15)) * Nn + nbase + nk,
        c0, c0 + 16);
#pragma unroll
    for (int j = 0; j < 6; ++j) {
      const int d = g * 96 + j * 16 + (lane & 15);
      v16bf bfr = load_frag(&ldsT[d * 40], c0, c0 + 16);
      acc[j] = wmma_bf16(a, bfr, acc[j]);
    }
  }

  const int hi = (lane >> 4) * 8;
#pragma unroll
  for (int j = 0; j < 6; ++j) {
    const int d = g * 96 + j * 16 + (lane & 15);
#pragma unroll
    for (int v = 0; v < 8; ++v) {
      const int m = r * 16 + v + hi;
      atomicAdd(&qtok[(size_t)(bh * Mm + m) * 192 + d], acc[j][v]);
    }
  }
}

// ---------------------------------------------------------------------------
// Per-(b,h) slice self-attention (64x64x64, ~0.1 GF total -> VALU is fine).
// Normalizes pooled qkv by slice_norm, softmax(q k^T * 0.125) @ v, and writes
// out_tok transposed ([d][m], bf16) so the scatter GEMM's B-fragments are
// K(=m)-contiguous.
// ---------------------------------------------------------------------------
__global__ void k_slice_attn(const float* __restrict__ qtok,
                             const float* __restrict__ snorm,
                             unsigned short* __restrict__ otokt) {
  __shared__ float qt[64 * 192];
  __shared__ float S[64 * 64];
  const int bh = blockIdx.x, tid = threadIdx.x;

  for (int idx = tid; idx < 64 * 192; idx += 128) {
    const int m = idx / 192;
    qt[idx] = qtok[(size_t)bh * (64 * 192) + idx] / (snorm[bh * Mm + m] + 1e-5f);
  }
  __syncthreads();

  for (int idx = tid; idx < 4096; idx += 128) {
    const int i = idx >> 6, j = idx & 63;
    float s = 0.0f;
    for (int d = 0; d < 64; ++d) s += qt[i * 192 + d] * qt[j * 192 + 64 + d];
    S[idx] = s * 0.125f;  // 1/sqrt(64)
  }
  __syncthreads();

  if (tid < 64) {
    float mx = -3.0e38f;
    for (int j = 0; j < 64; ++j) mx = fmaxf(mx, S[tid * 64 + j]);
    float sum = 0.0f;
    for (int j = 0; j < 64; ++j) {
      const float e = __expf(S[tid * 64 + j] - mx);
      S[tid * 64 + j] = e;
      sum += e;
    }
    const float inv = 1.0f / sum;
    for (int j = 0; j < 64; ++j) S[tid * 64 + j] *= inv;
  }
  __syncthreads();

  for (int idx = tid; idx < 4096; idx += 128) {
    const int i = idx >> 6, d = idx & 63;
    float o = 0.0f;
    for (int j = 0; j < 64; ++j) o += S[i * 64 + j] * qt[j * 192 + 128 + d];
    otokt[(size_t)(bh * 64 + d) * 64 + i] = f2bf(o);  // transposed store
  }
}

// ---------------------------------------------------------------------------
// Scatter back: token_out[b,n, h*64+d] = sum_m out_tok[m,d] * weights[m,n].
// Grid (N/64, H, B), 128 threads. Weight 64x64 tile is transposed into LDS
// once (A-fragments need [n][m]); B-fragments come straight from otokt [d][m].
// ---------------------------------------------------------------------------
__global__ void k_scatter(const unsigned short* __restrict__ wts,
                          const unsigned short* __restrict__ otokt,
                          unsigned short* __restrict__ tokout) {
  __shared__ unsigned short Wt[64 * 72];  // [n_local][m], padded
  const int n0 = blockIdx.x * 64, h = blockIdx.y, b = blockIdx.z;
  const int bh = b * Hh + h;
  const int tid = threadIdx.x, wave = tid >> 5, lane = tid & 31;

  {
    const int m = tid >> 1, nh = (tid & 1) * 32;
    const unsigned short* src = wts + (size_t)(bh * Mm + m) * Nn + n0 + nh;
#pragma unroll
    for (int c = 0; c < 4; ++c) {
      uint4 u = *(const uint4*)(src + c * 8);
      const unsigned short* p = (const unsigned short*)&u;
#pragma unroll
      for (int j = 0; j < 8; ++j) Wt[(nh + c * 8 + j) * 72 + m] = p[j];
    }
  }
  __syncthreads();

  const int c0 = (lane >> 4) * 8, c0b = (lane >> 4) * 16;
  v8f acc[4];
#pragma unroll
  for (int j = 0; j < 4; ++j) acc[j] = zero8();

#pragma unroll
  for (int k0 = 0; k0 < 64; k0 += 32) {
    v16bf a = load_frag(&Wt[(wave * 16 + (lane & 15)) * 72 + k0], c0, c0 + 16);
#pragma unroll
    for (int j = 0; j < 4; ++j) {
      v16bf bfr = load_frag(
          otokt + (size_t)(bh * 64 + j * 16 + (lane & 15)) * 64 + k0,
          c0, c0 + 16);
      acc[j] = wmma_bf16(a, bfr, acc[j]);
    }
  }

  const int hi = (lane >> 4) * 8;
#pragma unroll
  for (int j = 0; j < 4; ++j) {
    const int d = j * 16 + (lane & 15);
#pragma unroll
    for (int v = 0; v < 8; ++v) {
      const int n = n0 + wave * 16 + v + hi;
      tokout[(size_t)(b * Nn + n) * Dd + h * 64 + d] = f2bf(acc[j][v]);
    }
  }
}

// ---------------------------------------------------------------------------
// host launcher
// ---------------------------------------------------------------------------
extern "C" void kernel_launch(void* const* d_in, const int* in_sizes, int n_in,
                              void* d_out, int out_size, void* d_ws,
                              size_t ws_size, hipStream_t stream) {
  (void)in_sizes; (void)n_in; (void)out_size; (void)ws_size;

  const float* x    = (const float*)d_in[0];
  const float* Wk   = (const float*)d_in[1];
  const float* bk   = (const float*)d_in[2];
  const float* wtq  = (const float*)d_in[3];
  const float* temp = (const float*)d_in[4];
  const float* Wqkv = (const float*)d_in[5];
  const float* bqkv = (const float*)d_in[6];
  const float* Wo   = (const float*)d_in[7];
  const float* bo   = (const float*)d_in[8];
  float* out = (float*)d_out;

  char* cur = (char*)d_ws;
  auto alloc = [&](size_t bytes) -> void* {
    void* p = (void*)cur;
    cur += (bytes + 255) & ~(size_t)255;
    return p;
  };
  const size_t BN = (size_t)Bb * Nn;

  unsigned short* xb     = (unsigned short*)alloc(BN * Dd * 2);
  unsigned short* wkb    = (unsigned short*)alloc((size_t)Dd * Dd * 2);
  unsigned short* wqkvb  = (unsigned short*)alloc((size_t)3 * Dd * Dd * 2);
  unsigned short* wob    = (unsigned short*)alloc((size_t)Dd * Dd * 2);
  unsigned short* wtqb   = (unsigned short*)alloc((size_t)Hh * Mm * HD * 2);
  unsigned short* wtkb   = (unsigned short*)alloc(BN * Dd * 2);
  unsigned short* qkvb   = (unsigned short*)alloc(BN * 3 * Dd * 2);
  unsigned short* wtsb   = (unsigned short*)alloc((size_t)Bb * Hh * Mm * Nn * 2);
  float*          snorm  = (float*)alloc((size_t)Bb * Hh * Mm * 4);
  float*          qtok   = (float*)alloc((size_t)Bb * Hh * Mm * 192 * 4);
  unsigned short* otokt  = (unsigned short*)alloc((size_t)Bb * Hh * HD * Mm * 2);
  unsigned short* tokout = (unsigned short*)alloc(BN * Dd * 2);

  auto cvt = [&](const float* src, unsigned short* dst, int n) {
    k_f32_to_bf16<<<(n + 255) / 256, 256, 0, stream>>>(src, dst, n);
  };
  cvt(x, xb, (int)(BN * Dd));
  cvt(Wk, wkb, Dd * Dd);
  cvt(Wqkv, wqkvb, 3 * Dd * Dd);
  cvt(Wo, wob, Dd * Dd);
  cvt(wtq, wtqb, Hh * Mm * HD);
  {
    const int n = Bb * Hh * Mm * 192;
    k_zero_f32<<<(n + 255) / 256, 256, 0, stream>>>(qtok, n);
  }

  // wtk = x @ Wk^T + bk   (bf16 out)
  k_gemm<true><<<dim3(Dd / 64, (unsigned)(BN / 128)), 256, 0, stream>>>(
      xb, wkb, bk, wtkb, Dd, Dd);
  // qkv = x @ Wqkv^T + bqkv  (bf16 out)
  k_gemm<true><<<dim3(3 * Dd / 64, (unsigned)(BN / 128)), 256, 0, stream>>>(
      xb, wqkvb, bqkv, qkvb, 3 * Dd, Dd);
  // slice scores + softmax over M -> bf16 weights
  k_scores_softmax<<<dim3(Nn / 16, Hh, Bb), 32, 0, stream>>>(wtqb, wtkb, temp,
                                                             wtsb);
  // slice_norm[b,h,m]
  k_slice_norm<<<Bb * Hh * Mm, 256, 0, stream>>>(wtsb, snorm);
  // pooled qkv tokens (f32 atomic accumulate across 8 K-chunks)
  k_pool_qkv<<<dim3(Bb * Hh, 8), 256, 0, stream>>>(wtsb, qkvb, qtok);
  // tiny slice self-attention
  k_slice_attn<<<Bb * Hh, 128, 0, stream>>>(qtok, snorm, otokt);
  // scatter back to tokens
  k_scatter<<<dim3(Nn / 64, Hh, Bb), 128, 0, stream>>>(wtsb, otokt, tokout);
  // final projection: out = token_out @ Wo^T + bo  (f32 out)
  k_gemm<false><<<dim3(Dd / 64, (unsigned)(BN / 128)), 256, 0, stream>>>(
      tokout, wob, bo, out, Dd, Dd);
}
